// RWKV_TimeMix_29609504539184
// MI455X (gfx1250) — matrix operand order repaired and checked by reading, not compiled, not run on hardware
//
#include <hip/hip_runtime.h>

// ---------------------------------------------------------------------------
// RWKV TimeMix for MI455X (gfx1250, wave32). Fully fused: one workgroup per
// batch row walks T=200 in 13 chunks of 16 steps. All GEMMs are
// v_wmma_f32_16x16x32_bf16; the O(T^2) decay einsum is a first-order linear
// recurrence carried in registers; alpha/beta tables staged via the Tensor
// Data Mover (tensor_load_to_lds) when the builtin is available.
// Eliminates ~1.26 GB of HBM intermediate traffic vs the unfused pipeline.
// ---------------------------------------------------------------------------

typedef __attribute__((ext_vector_type(16))) __bf16 v16bf;
typedef __attribute__((ext_vector_type(8)))  float  v8f;

static constexpr int kB = 2048;
static constexpr int kT = 200;
static constexpr int kC = 128;   // HID
static constexpr int kA = 128;   // ATTN
static constexpr int kNH = 8;
static constexpr int kChunks = (kT + 15) / 16;   // 13

union BFrag {
    v16bf v;
    uint4 q[2];
    unsigned short s[16];
};

__device__ __forceinline__ unsigned short f2bf(float f) {
    unsigned u = __float_as_uint(f);
    u += 0x7FFFu + ((u >> 16) & 1u);       // round-to-nearest-even
    return (unsigned short)(u >> 16);
}

// ---------------------------------------------------------------------------
// Kernel 0: convert the four 128x128 f32 weight matrices to bf16 (row-major
// [out][in]; each WMMA B-fragment is then two 16B loads per lane, 32B apart).
// ---------------------------------------------------------------------------
__global__ __launch_bounds__(256) void convert_w_kernel(
    const float* __restrict__ Wk, const float* __restrict__ Wv,
    const float* __restrict__ Wr, const float* __restrict__ Wo,
    unsigned short* __restrict__ wk, unsigned short* __restrict__ wv,
    unsigned short* __restrict__ wr, unsigned short* __restrict__ wo)
{
    int idx = blockIdx.x * 256 + threadIdx.x;
    if (idx < kA * kC) {
        wk[idx] = f2bf(Wk[idx]);
        wv[idx] = f2bf(Wv[idx]);
        wr[idx] = f2bf(Wr[idx]);
        wo[idx] = f2bf(Wo[idx]);
    }
}

// ---------------------------------------------------------------------------
// Tensor Data Mover: 1-D f32 table -> LDS. D# built per ISA 8.3/8.4 bitfields.
// 6-arg builtin form: (uint32x4 g0, int32x8 g1, int32x4 g2, int32x4 g3,
//                      int32x8 extra, i32 cpol)
// ---------------------------------------------------------------------------
__device__ __forceinline__ unsigned lds_offset_of(const void* p) {
    return (unsigned)(unsigned long long)
        (__attribute__((address_space(3))) const char*)p;
}

#if __has_builtin(__builtin_amdgcn_tensor_load_to_lds)
typedef __attribute__((ext_vector_type(4))) unsigned int v4u;
typedef __attribute__((ext_vector_type(8))) int          v8i_t;
typedef __attribute__((ext_vector_type(4))) int          v4i_t;

__device__ __forceinline__ void tdm_load_f32_1d(const void* gptr,
                                                unsigned lds_off,
                                                unsigned nelem) {
    unsigned long long ga = (unsigned long long)gptr;
    v4u g0;
    g0.x = 1u;                                           // count=1 (user D#)
    g0.y = lds_off;                                      // lds_addr [63:32]
    g0.z = (unsigned)(ga & 0xFFFFFFFFu);                 // global_addr[31:0]
    g0.w = (unsigned)((ga >> 32) & 0x01FFFFFFu)          // global_addr[56:32]
         | (2u << 30);                                   // type = 2 ("image")
    v8i_t g1 = {};
    g1[0] = (int)(2u << 16);                             // data_size=4B, wg_mask=0
    g1[1] = (int)((nelem & 0xFFFFu) << 16);              // tensor_dim0[15:0]
    g1[2] = (int)(((nelem >> 16) & 0xFFFFu)              // tensor_dim0[31:16]
         | (1u << 16));                                  // tensor_dim1 = 1
    g1[3] = (int)((nelem & 0xFFFFu) << 16);              // tile_dim0 = nelem
    g1[4] = 0;                                           // tile_dim1/2 unused
    g1[5] = (int)nelem;                                  // tensor_dim0_stride
    g1[6] = 0; g1[7] = 0;
    v4i_t z4 = {};
    v8i_t z8 = {};
    __builtin_amdgcn_tensor_load_to_lds(g0, g1, z4, z4, z8, 0);
}
#endif

// ---------------------------------------------------------------------------
// Fused kernel: one block per batch row b.
// ---------------------------------------------------------------------------
__global__ __launch_bounds__(256) void fused_timemix_kernel(
    const int* __restrict__ log_seqs, const float* __restrict__ emb,
    const unsigned short* __restrict__ wk, const unsigned short* __restrict__ wv,
    const unsigned short* __restrict__ wr, const unsigned short* __restrict__ wo,
    const float* __restrict__ bk, const float* __restrict__ bv,
    const float* __restrict__ br, const float* __restrict__ bo,
    const float* __restrict__ time_w, const float* __restrict__ alpha,
    const float* __restrict__ beta,   const float* __restrict__ gamma,
    const int* __restrict__ pos_seqs, const int* __restrict__ neg_seqs,
    float* __restrict__ outPos, float* __restrict__ outNeg)
{
    __shared__ __align__(16) unsigned short xs[16 * 136];   // x chunk (bf16)
    __shared__ float kk[16 * 129];                          // k chunk (f32)
    __shared__ float vv[16 * 129];                          // v chunk (f32)
    __shared__ float rr[16 * 129];                          // r chunk (f32)
    __shared__ __align__(16) unsigned short rw[16 * 136];   // rwkv chunk (bf16)
    __shared__ float a_lds[kNH * kT];                       // alpha table
    __shared__ float b_lds[kNH * kT];                       // beta table
    __shared__ float negs[16];

    const int b    = blockIdx.x;
    const int tid  = threadIdx.x;
    const int wave = tid >> 5, lane = tid & 31;
    const int l15  = lane & 15, half = lane >> 4;
    const int n0   = wave * 16;
    const int* seq = log_seqs + b * kT;

    // ---- stage alpha/beta into LDS (TDM when available) --------------------
#if __has_builtin(__builtin_amdgcn_tensor_load_to_lds)
    if (wave == 0) {
        tdm_load_f32_1d(alpha, lds_offset_of(a_lds), kNH * kT);
        tdm_load_f32_1d(beta,  lds_offset_of(b_lds), kNH * kT);
    }
#if __has_builtin(__builtin_amdgcn_s_wait_tensorcnt)
    __builtin_amdgcn_s_wait_tensorcnt(0);
#endif
    __syncthreads();
#else
    for (int i = tid; i < kNH * kT; i += 256) {
        a_lds[i] = alpha[i];
        b_lds[i] = beta[i];
    }
    __syncthreads();
#endif

    // ---- per-channel recurrence state (threads 0..127 own channel tid) -----
    float sk_state = 0.f, sw_state = 0.f, decay = 0.f;
    if (tid < kA) decay = time_w[(tid >> 4) * kT + (kT - 2)]; // exp(-ds_h)

    for (int chunk = 0; chunk < kChunks; ++chunk) {
        const int t0 = chunk * 16;

        // ==== 1) gather + time-shift x tile into LDS (bf16) ================
        {
            const int row = tid >> 4;
            const int c0  = (tid & 15) * 8;   // 8-ch run, never crosses ch 64
            const int t   = t0 + row;
            int src = -1;
            if (t < kT) {
                if (c0 < kC / 2) { if (t > 0) src = seq[t - 1]; }
                else               src = seq[t];
            }
            float4 f0 = make_float4(0.f, 0.f, 0.f, 0.f), f1 = f0;
            if (src >= 0) {
                const float4* p = (const float4*)(emb + (size_t)src * kC + c0);
                f0 = p[0]; f1 = p[1];
            }
            unsigned short* d = &xs[row * 136 + c0];
            d[0] = f2bf(f0.x); d[1] = f2bf(f0.y); d[2] = f2bf(f0.z); d[3] = f2bf(f0.w);
            d[4] = f2bf(f1.x); d[5] = f2bf(f1.y); d[6] = f2bf(f1.z); d[7] = f2bf(f1.w);
            // prefetch next chunk's embedding rows (global_prefetch_b8)
            const int t2 = t0 + 16 + row;
            if (t2 < kT) {
                const int s2 = seq[t2];
                __builtin_prefetch(emb + (size_t)s2 * kC + c0, 0, 0);
            }
        }
        __syncthreads();

        // ==== 2) k/v/r projections: 3 x (4 WMMA) per wave ==================
        v8f ck = {}, cv = {}, cr = {};
        for (int kt = 0; kt < 4; ++kt) {
            const int k0 = kt * 32;
            BFrag a, bK, bV, bR;
            const unsigned short* ap = &xs[l15 * 136 + k0 + 8 * half];
            a.q[0] = *(const uint4*)ap;
            a.q[1] = *(const uint4*)(ap + 16);
            const int wrow = (n0 + l15) * kC + k0 + 8 * half;
            bK.q[0] = *(const uint4*)&wk[wrow]; bK.q[1] = *(const uint4*)&wk[wrow + 16];
            bV.q[0] = *(const uint4*)&wv[wrow]; bV.q[1] = *(const uint4*)&wv[wrow + 16];
            bR.q[0] = *(const uint4*)&wr[wrow]; bR.q[1] = *(const uint4*)&wr[wrow + 16];
            ck = __builtin_amdgcn_wmma_f32_16x16x32_bf16(false, a.v, false, bK.v, (short)0, ck, false, false);
            cv = __builtin_amdgcn_wmma_f32_16x16x32_bf16(false, a.v, false, bV.v, (short)0, cv, false, false);
            cr = __builtin_amdgcn_wmma_f32_16x16x32_bf16(false, a.v, false, bR.v, (short)0, cr, false, false);
        }
        {
            const int g = n0 + l15;
            const float bkv = bk[g], bvv = bv[g], brv = br[g];
            #pragma unroll
            for (int i = 0; i < 8; ++i) {
                const int rowi = i + 8 * half;        // D row = i + 8*(lane/16)
                float kx = ck[i] + bkv;
                kx = fminf(fmaxf(kx, -60.f), 30.f);
                kk[rowi * 129 + g] = expf(kx);
                vv[rowi * 129 + g] = cv[i] + bvv;
                rr[rowi * 129 + g] = cr[i] + brv;
            }
        }
        __syncthreads();

        // ==== 3) 16-step decay recurrence + cumsum + sigmoid gate ==========
        if (tid < kA) {
            const int a = tid, h = a >> 4;
            #pragma unroll
            for (int i = 0; i < 16; ++i) {
                const int t  = t0 + i;
                const int tc = (t < kT) ? t : (kT - 1);
                const float k = kk[i * 129 + a];
                const float v = vv[i * 129 + a];
                const float r = rr[i * 129 + a];
                sk_state += k;
                sw_state = decay * sw_state + k * v * a_lds[h * kT + tc];
                const float wkv = b_lds[h * kT + tc] * sw_state;
                const float sig = 1.f / (1.f + expf(-r));
                rw[i * 136 + a] = f2bf(sig * wkv / sk_state);  // sk>0 always
            }
        }
        if (tid < 16) negs[tid] = 0.f;
        __syncthreads();

        // ==== 4) output projection + pos/neg logits ========================
        v8f acc = {};
        for (int kt = 0; kt < 4; ++kt) {
            const int k0 = kt * 32;
            BFrag a, bF;
            const unsigned short* ap = &rw[l15 * 136 + k0 + 8 * half];
            a.q[0] = *(const uint4*)ap;
            a.q[1] = *(const uint4*)(ap + 16);
            const int wrow = (n0 + l15) * kA + k0 + 8 * half;
            bF.q[0] = *(const uint4*)&wo[wrow];
            bF.q[1] = *(const uint4*)&wo[wrow + 16];
            acc = __builtin_amdgcn_wmma_f32_16x16x32_bf16(false, a.v, false, bF.v, (short)0, acc, false, false);
        }
        {
            const int g = n0 + l15;
            const float bog = bo[g];
            #pragma unroll
            for (int i = 0; i < 8; ++i) {
                const int t = t0 + i + 8 * half;      // uniform within half-wave
                float np = 0.f;
                if (t < kT) {
                    const size_t m = (size_t)b * kT + t;
                    const float lf = (acc[i] + bog) * gamma[t];
                    const int pi = pos_seqs[m];
                    const int ni = neg_seqs[m];
                    outPos[m * kC + g] = lf * emb[(size_t)pi * kC + g];
                    np = lf * emb[(size_t)ni * kC + g];
                }
                #pragma unroll
                for (int s = 1; s < 16; s <<= 1) np += __shfl_xor(np, s, 16);
                if (l15 == 0 && t < kT) atomicAdd(&negs[i + 8 * half], np);
            }
        }
        __syncthreads();
        if (tid < 16) {
            const int t = t0 + tid;
            if (t < kT) outNeg[(size_t)b * kT + t] = negs[tid];
        }
        __syncthreads();
    }
}

// ---------------------------------------------------------------------------
extern "C" void kernel_launch(void* const* d_in, const int* in_sizes, int n_in,
                              void* d_out, int out_size, void* d_ws, size_t ws_size,
                              hipStream_t stream) {
    (void)in_sizes; (void)n_in; (void)out_size; (void)ws_size;
    const int*   log_seqs = (const int*)d_in[1];
    const int*   pos_seqs = (const int*)d_in[2];
    const int*   neg_seqs = (const int*)d_in[3];
    const float* emb      = (const float*)d_in[4];
    const float* time_w   = (const float*)d_in[5];
    const float* alpha    = (const float*)d_in[6];
    const float* beta     = (const float*)d_in[7];
    const float* gamma    = (const float*)d_in[8];
    const float* Wk = (const float*)d_in[9];  const float* bk = (const float*)d_in[10];
    const float* Wv = (const float*)d_in[11]; const float* bv = (const float*)d_in[12];
    const float* Wr = (const float*)d_in[13]; const float* br = (const float*)d_in[14];
    const float* Wo = (const float*)d_in[15]; const float* bo = (const float*)d_in[16];

    // workspace: just the four bf16 weight matrices (128 KB)
    unsigned short* wsWk = (unsigned short*)d_ws;
    unsigned short* wsWv = wsWk + kA * kC;
    unsigned short* wsWr = wsWv + kA * kC;
    unsigned short* wsWo = wsWr + kA * kC;

    float* outPos = (float*)d_out;
    float* outNeg = outPos + (size_t)kB * kT * kC;

    convert_w_kernel<<<(kA * kC + 255) / 256, 256, 0, stream>>>(
        Wk, Wv, Wr, Wo, wsWk, wsWv, wsWr, wsWo);

    fused_timemix_kernel<<<kB, 256, 0, stream>>>(
        log_seqs, emb, wsWk, wsWv, wsWr, wsWo, bk, bv, br, bo,
        time_w, alpha, beta, gamma, pos_seqs, neg_seqs, outPos, outNeg);
}
